// RandLAUnit_68496138437088
// MI455X (gfx1250) — compile-verified, matrix operand order checked by reading, not run on hardware
//
#include <hip/hip_runtime.h>

typedef __attribute__((ext_vector_type(16))) _Float16 v16h;
typedef __attribute__((ext_vector_type(8)))  float    v8f;

#define BB 2
#define NN 16384
#define MM 4096
#define CC 32
#define DD 64
#define HH 32
#define KK 16
#define EPSS 1e-5f
#define QPB 8     // queries (waves) per KNN block
#define TS  512   // KNN point tile (fits 6KB LDS)

// ---------------------------------------------------------------------------
// prep: zero stat accumulators, convert weights to f16 for WMMA
// ---------------------------------------------------------------------------
__global__ void prep_kernel(const float* __restrict__ Ws, const float* __restrict__ Wo,
                            const float* __restrict__ W1, const float* __restrict__ W2,
                            _Float16* __restrict__ hs, _Float16* __restrict__ ho,
                            _Float16* __restrict__ h1, _Float16* __restrict__ h2,
                            float* __restrict__ statsU, float* __restrict__ stats3) {
  int i = blockIdx.x * blockDim.x + threadIdx.x;
  if (i < 128)  statsU[i] = 0.f;
  if (i < 384)  stats3[i] = 0.f;
  if (i < 4096) { hs[i] = (_Float16)Ws[i]; ho[i] = (_Float16)Wo[i]; }
  if (i < 2048) { h1[i] = (_Float16)W1[i]; h2[i] = (_Float16)W2[i]; }
}

// ---------------------------------------------------------------------------
// KNN: one wave per query. LDS point tiles (+ prefetch of the next tile),
// per-lane register top-16 (u64 keys = dist bits | idx), wave-wide bitonic
// top-k merge via shfl_xor. Also emits xyz_ds and sample_idx outputs.
// ---------------------------------------------------------------------------
__global__ void knn_kernel(const float* __restrict__ xyz, const int* __restrict__ sampleIdx,
                           int* __restrict__ knnIdx, float* __restrict__ outXyzDs,
                           int* __restrict__ outSidx) {
  __shared__ float tile[TS * 3];
  int tid  = threadIdx.x;
  int wave = tid >> 5, lane = tid & 31;
  int bm = blockIdx.x * QPB + wave;          // M/QPB divides evenly; block is single-batch
  int b  = bm / MM;
  int sidx = sampleIdx[bm];
  const float* xb = xyz + (size_t)b * NN * 3;
  float qx = xb[sidx * 3 + 0], qy = xb[sidx * 3 + 1], qz = xb[sidx * 3 + 2];
  if (lane == 0) {
    outXyzDs[bm * 3 + 0] = qx; outXyzDs[bm * 3 + 1] = qy; outXyzDs[bm * 3 + 2] = qz;
    outSidx[bm] = sidx;
  }

  unsigned long long h[16];
#pragma unroll
  for (int i = 0; i < 16; ++i) h[i] = 0xFFFFFFFF00000000ULL + (unsigned)i;  // distinct sentinels
  unsigned long long worst = h[15];

  for (int t = 0; t < NN; t += TS) {
    __syncthreads();
    for (int i = tid; i < TS * 3; i += blockDim.x) tile[i] = xb[t * 3 + i];
    __syncthreads();
    // hide next tile's HBM/L2 latency behind the scan (global_prefetch_b8)
    if (t + TS < NN)
      __builtin_prefetch(xb + (size_t)(t + TS) * 3 + tid * 6, 0, 1);
#pragma unroll
    for (int j = 0; j < TS / 32; ++j) {
      int li = lane + 32 * j;
      float dx = qx - tile[li * 3 + 0];
      float dy = qy - tile[li * 3 + 1];
      float dz = qz - tile[li * 3 + 2];
      float d2 = dx * dx + dy * dy + dz * dz;
      unsigned long long key =
          ((unsigned long long)__float_as_uint(d2) << 32) | (unsigned)(t + li);
      if (key < worst) {
#pragma unroll
        for (int i = 0; i < 16; ++i) if (h[i] == worst) h[i] = key;  // keys unique
        worst = h[0];
#pragma unroll
        for (int i = 1; i < 16; ++i) worst = (h[i] > worst) ? h[i] : worst;
      }
    }
  }

  // per-lane bitonic sort ascending (constant indices -> stays in VGPRs)
#pragma unroll
  for (int ksz = 2; ksz <= 16; ksz <<= 1) {
#pragma unroll
    for (int jsz = ksz >> 1; jsz > 0; jsz >>= 1) {
#pragma unroll
      for (int i = 0; i < 16; ++i) {
        int l = i ^ jsz;
        if (l > i) {
          bool up = ((i & ksz) == 0);
          bool sw = up ? (h[i] > h[l]) : (h[i] < h[l]);
          if (sw) { unsigned long long tmp = h[i]; h[i] = h[l]; h[l] = tmp; }
        }
      }
    }
  }

  // butterfly merges: keep smallest 16 of union at every stage; all lanes converge
#pragma unroll
  for (int s = 1; s <= 16; s <<= 1) {
    unsigned long long o[16];
#pragma unroll
    for (int i = 0; i < 16; ++i) o[i] = __shfl_xor(h[15 - i], s, 32);
#pragma unroll
    for (int i = 0; i < 16; ++i) if (o[i] < h[i]) h[i] = o[i];   // now bitonic
#pragma unroll
    for (int jsz = 8; jsz > 0; jsz >>= 1) {
#pragma unroll
      for (int i = 0; i < 16; ++i) {
        int l = i ^ jsz;
        if (l > i && h[i] > h[l]) { unsigned long long tmp = h[i]; h[i] = h[l]; h[l] = tmp; }
      }
    }
  }

  if (lane == 0) {
#pragma unroll
    for (int i = 0; i < 16; ++i)
      knnIdx[bm * KK + i] = (int)(h[i] & 0xFFFFFFFFULL);
  }
}

// ---------------------------------------------------------------------------
// stage1: per (b,m,k) row -> pos-enc MLP + feature MLP pre-BN outputs U[.,64],
// plus per-channel sum/sumsq (LDS reduce -> global atomics)
// ---------------------------------------------------------------------------
__global__ void stage1_kernel(const float* __restrict__ xyz, const float* __restrict__ feats,
                              const int* __restrict__ sampleIdx, const int* __restrict__ knnIdx,
                              const float* __restrict__ W_pos, const float* __restrict__ b_pos,
                              const float* __restrict__ W_feat, const float* __restrict__ b_feat,
                              float* __restrict__ U, float* __restrict__ statsU) {
  __shared__ float sWf[CC * HH];
  __shared__ float sWp[10 * HH];
  __shared__ float sbf[HH], sbp[HH];
  __shared__ float sacc[128];
  int tid = threadIdx.x;
  for (int i = tid; i < CC * HH; i += blockDim.x) sWf[i] = W_feat[i];
  for (int i = tid; i < 10 * HH; i += blockDim.x) sWp[i] = W_pos[i];
  if (tid < HH) { sbf[tid] = b_feat[tid]; sbp[tid] = b_pos[tid]; }
  if (tid < 128) sacc[tid] = 0.f;
  __syncthreads();

  int row = blockIdx.x * blockDim.x + tid;   // [0, B*M*K)
  int bm = row / KK;
  int b  = bm / MM;
  int sidx = sampleIdx[bm];
  int nidx = knnIdx[row];
  const float* xb = xyz + (size_t)b * NN * 3;
  float qx = xb[sidx * 3], qy = xb[sidx * 3 + 1], qz = xb[sidx * 3 + 2];
  float px = xb[nidx * 3], py = xb[nidx * 3 + 1], pz = xb[nidx * 3 + 2];
  float rx = qx - px, ry = qy - py, rz = qz - pz;
  float dist = sqrtf(rx * rx + ry * ry + rz * rz);
  float pf[10] = {qx, qy, qz, px, py, pz, rx, ry, rz, dist};
  float f[CC];
#pragma unroll
  for (int c = 0; c < CC; ++c) f[c] = feats[((size_t)b * CC + c) * NN + nidx];

  size_t ro = (size_t)row * DD;
#pragma unroll 4
  for (int hh = 0; hh < HH; ++hh) {
    float uf = sbf[hh];
#pragma unroll
    for (int c = 0; c < CC; ++c) uf += f[c] * sWf[c * HH + hh];
    float up = sbp[hh];
#pragma unroll
    for (int i = 0; i < 10; ++i) up += pf[i] * sWp[i * HH + hh];
    U[ro + hh]      = uf;     // combined channels [0,32) = feat half
    U[ro + HH + hh] = up;     // combined channels [32,64) = pos half
    atomicAdd(&sacc[hh], uf);           atomicAdd(&sacc[64 + hh], uf * uf);
    atomicAdd(&sacc[HH + hh], up);      atomicAdd(&sacc[64 + HH + hh], up * up);
  }
  __syncthreads();
  if (tid < 128) atomicAdd(&statsU[tid], sacc[tid]);
}

// ---------------------------------------------------------------------------
// neighbor-mean features (for sc1) and center-feature gather (for sc2)
// ---------------------------------------------------------------------------
__global__ void scmean_kernel(const float* __restrict__ feats, const int* __restrict__ sampleIdx,
                              const int* __restrict__ knnIdx,
                              float* __restrict__ meanF, float* __restrict__ centF) {
  int idx = blockIdx.x * blockDim.x + threadIdx.x;  // [0, B*M*C)
  int c = idx % CC; int bm = idx / CC; int b = bm / MM;
  const float* fb = feats + ((size_t)b * CC + c) * NN;
  float s = 0.f;
#pragma unroll
  for (int k = 0; k < KK; ++k) s += fb[knnIdx[bm * KK + k]];
  meanF[bm * CC + c] = s * (1.f / KK);
  centF[bm * CC + c] = fb[sampleIdx[bm]];
}

// ---------------------------------------------------------------------------
// finalize BN stats for U -> per-channel scale/shift
// ---------------------------------------------------------------------------
__global__ void finalizeU_kernel(const float* __restrict__ statsU,
                                 const float* __restrict__ g_feat, const float* __restrict__ be_feat,
                                 const float* __restrict__ g_pos,  const float* __restrict__ be_pos,
                                 float* __restrict__ bnU) {
  int c = threadIdx.x; if (c >= 64) return;
  float cnt  = (float)((size_t)BB * MM * KK);
  float mean = statsU[c] / cnt;
  float var  = statsU[64 + c] / cnt - mean * mean;
  float inv  = rsqrtf(var + EPSS);
  float g  = (c < HH) ? g_feat[c]  : g_pos[c - HH];
  float be = (c < HH) ? be_feat[c] : be_pos[c - HH];
  bnU[c]      = g * inv;
  bnU[64 + c] = be - g * mean * inv;
}

// ---------------------------------------------------------------------------
// attention: one wave per (b,m). U tile staged memory->LDS with
// GLOBAL_LOAD_ASYNC_TO_LDS_B128 (ASYNCcnt path, no VGPR round-trip),
// BN+ReLU applied in place, then WMMA scores = combined[16x64] @ W[64x64],
// softmax over K via shfl_xor(16) on the D-fragment layout, pool -> pooled.
// ---------------------------------------------------------------------------
__global__ void attn_kernel(const float* __restrict__ U, const float* __restrict__ bnU,
                            const _Float16* __restrict__ Wh, float* __restrict__ pooled) {
  __shared__ float comb[4][KK][DD];
  int tid = threadIdx.x; int w = tid >> 5; int lane = tid & 31;
  int bm = blockIdx.x * 4 + w;
  const float* Ub = U + (size_t)bm * KK * DD;

  // async stage: 4KB tile, 8 x b128 per lane (32 lanes x 16B x 8 = 4096B)
  unsigned ldsBase = (unsigned)(size_t)(&comb[w][0][0]);   // LDS aperture: offset in addr[31:0]
  {
    unsigned long long ga = (unsigned long long)(const char*)Ub + (unsigned)(lane * 16);
    unsigned la = ldsBase + (unsigned)(lane * 16);
#pragma unroll
    for (int it = 0; it < 8; ++it) {
      asm volatile("global_load_async_to_lds_b128 %0, %1, off"
                   :: "v"(la), "v"(ga) : "memory");
      la += 512; ga += 512;
    }
  }
  asm volatile("s_wait_asynccnt 0x0" ::: "memory");
  __syncthreads();

  // BN + ReLU in place in LDS
  for (int i = lane; i < KK * DD; i += 32) {
    int c = i & 63;
    float v = comb[w][i >> 6][c] * bnU[c] + bnU[64 + c];
    comb[w][i >> 6][c] = v > 0.f ? v : 0.f;
  }
  __syncthreads();

  int l = lane & 15, hi = (lane >> 4) & 1;
  v16h a[2];
#pragma unroll
  for (int kb = 0; kb < 2; ++kb) {
    int kbase = kb * 32 + hi * 8;
#pragma unroll
    for (int j = 0; j < 8; ++j) {
      a[kb][j]     = (_Float16)comb[w][l][kbase + j];
      a[kb][8 + j] = (_Float16)comb[w][l][kbase + 16 + j];
    }
  }

#pragma unroll
  for (int nt = 0; nt < 4; ++nt) {
    v8f acc;
#pragma unroll
    for (int i = 0; i < 8; ++i) acc[i] = 0.f;
#pragma unroll
    for (int kb = 0; kb < 2; ++kb) {
      v16h bf;
      int kbase = kb * 32 + hi * 8;
#pragma unroll
      for (int j = 0; j < 8; ++j) {
        bf[j]     = Wh[(kbase + j) * DD + nt * 16 + l];
        bf[8 + j] = Wh[(kbase + 16 + j) * DD + nt * 16 + l];
      }
      acc = __builtin_amdgcn_wmma_f32_16x16x32_f16(false, a[kb], false, bf,
                                                   (short)0, acc, false, false);
    }
    // softmax over the 16 neighbor rows of this column (split lane <-> lane^16)
    float mx = acc[0];
#pragma unroll
    for (int v = 1; v < 8; ++v) mx = fmaxf(mx, acc[v]);
    mx = fmaxf(mx, __shfl_xor(mx, 16, 32));
    float e[8]; float sum = 0.f;
#pragma unroll
    for (int v = 0; v < 8; ++v) { e[v] = __expf(acc[v] - mx); sum += e[v]; }
    sum += __shfl_xor(sum, 16, 32);
    float invs = 1.f / sum;
    int col = nt * 16 + l;
    int rbase = hi * 8;
    float pp = 0.f;
#pragma unroll
    for (int v = 0; v < 8; ++v) pp += e[v] * invs * comb[w][rbase + v][col];
    pp += __shfl_xor(pp, 16, 32);
    if (lane < 16) pooled[(size_t)bm * DD + col] = pp;
  }
}

// ---------------------------------------------------------------------------
// generic 16-row-tile WMMA GEMM  Y[R,64] = X[R,KD] @ Wh[KD,64]  + BN stats
// ---------------------------------------------------------------------------
template <int KD>
__global__ void gemm_stats_kernel(const float* __restrict__ X, const _Float16* __restrict__ Wh,
                                  float* __restrict__ Y, float* __restrict__ stats) {
  int tid = threadIdx.x; int w = tid >> 5; int lane = tid & 31;
  int rowbase = (blockIdx.x * 4 + w) * 16;
  int l = lane & 15, hi = (lane >> 4) & 1;
  constexpr int NKB = KD / 32;
  v16h a[NKB];
#pragma unroll
  for (int kb = 0; kb < NKB; ++kb) {
    int kbase = kb * 32 + hi * 8;
    const float* xr = X + (size_t)(rowbase + l) * KD;
#pragma unroll
    for (int j = 0; j < 8; ++j) {
      a[kb][j]     = (_Float16)xr[kbase + j];
      a[kb][8 + j] = (_Float16)xr[kbase + 16 + j];
    }
  }
#pragma unroll
  for (int nt = 0; nt < 4; ++nt) {
    v8f acc;
#pragma unroll
    for (int i = 0; i < 8; ++i) acc[i] = 0.f;
#pragma unroll
    for (int kb = 0; kb < NKB; ++kb) {
      v16h bf;
      int kbase = kb * 32 + hi * 8;
#pragma unroll
      for (int j = 0; j < 8; ++j) {
        bf[j]     = Wh[(kbase + j) * DD + nt * 16 + l];
        bf[8 + j] = Wh[(kbase + 16 + j) * DD + nt * 16 + l];
      }
      acc = __builtin_amdgcn_wmma_f32_16x16x32_f16(false, a[kb], false, bf,
                                                   (short)0, acc, false, false);
    }
    int col = nt * 16 + l;
    float s = 0.f, q = 0.f;
#pragma unroll
    for (int v = 0; v < 8; ++v) {
      float y = acc[v];
      Y[(size_t)(rowbase + hi * 8 + v) * DD + col] = y;
      s += y; q += y * y;
    }
    s += __shfl_xor(s, 16, 32);
    q += __shfl_xor(q, 16, 32);
    if (lane < 16) { atomicAdd(&stats[col], s); atomicAdd(&stats[64 + col], q); }
  }
}

// ---------------------------------------------------------------------------
// finalize BN stats for out / sc1 / sc2
// ---------------------------------------------------------------------------
__global__ void finalize3_kernel(const float* __restrict__ stats3,
                                 const float* __restrict__ g_out, const float* __restrict__ be_out,
                                 const float* __restrict__ g_sc1, const float* __restrict__ be_sc1,
                                 const float* __restrict__ g_sc2, const float* __restrict__ be_sc2,
                                 float* __restrict__ bn3) {
  int t = threadIdx.x; if (t >= 192) return;
  int sec = t / 64, c = t % 64;
  const float* st = stats3 + sec * 128;
  float cnt  = (float)((size_t)BB * MM);
  float mean = st[c] / cnt;
  float var  = st[64 + c] / cnt - mean * mean;
  float inv  = rsqrtf(var + EPSS);
  float g  = (sec == 0) ? g_out[c]  : (sec == 1) ? g_sc1[c]  : g_sc2[c];
  float be = (sec == 0) ? be_out[c] : (sec == 1) ? be_sc1[c] : be_sc2[c];
  bn3[sec * 128 + c]      = g * inv;
  bn3[sec * 128 + 64 + c] = be - g * mean * inv;
}

// ---------------------------------------------------------------------------
// final: y = relu( relu(bn(O)) + bn(S1) + bn(S2) ), write transposed [B,D,M]
// ---------------------------------------------------------------------------
__global__ void final_kernel(const float* __restrict__ O, const float* __restrict__ S1,
                             const float* __restrict__ S2, const float* __restrict__ bn3,
                             float* __restrict__ outY) {
  int idx = blockIdx.x * blockDim.x + threadIdx.x;  // [0, B*M*D)
  int d = idx % DD; int bm = idx / DD; int b = bm / MM; int m = bm % MM;
  float o  = O[idx]  * bn3[d]        + bn3[64 + d];  o = o > 0.f ? o : 0.f;
  float s1 = S1[idx] * bn3[128 + d]  + bn3[192 + d];
  float s2 = S2[idx] * bn3[256 + d]  + bn3[320 + d];
  float y = o + s1 + s2;
  outY[((size_t)b * DD + d) * MM + m] = y > 0.f ? y : 0.f;
}

// ---------------------------------------------------------------------------
extern "C" void kernel_launch(void* const* d_in, const int* in_sizes, int n_in,
                              void* d_out, int out_size, void* d_ws, size_t ws_size,
                              hipStream_t stream) {
  (void)in_sizes; (void)n_in; (void)out_size; (void)ws_size;
  const float* xyz       = (const float*)d_in[0];
  const float* feats     = (const float*)d_in[1];
  const int*   sampleIdx = (const int*)d_in[2];
  const float* W_pos  = (const float*)d_in[3];
  const float* b_pos  = (const float*)d_in[4];
  const float* g_pos  = (const float*)d_in[5];
  const float* be_pos = (const float*)d_in[6];
  const float* W_feat  = (const float*)d_in[7];
  const float* b_feat  = (const float*)d_in[8];
  const float* g_feat  = (const float*)d_in[9];
  const float* be_feat = (const float*)d_in[10];
  const float* W_score = (const float*)d_in[11];
  const float* W_out  = (const float*)d_in[12];
  const float* g_out  = (const float*)d_in[13];
  const float* be_out = (const float*)d_in[14];
  const float* W_sc1  = (const float*)d_in[15];
  const float* g_sc1  = (const float*)d_in[16];
  const float* be_sc1 = (const float*)d_in[17];
  const float* W_sc2  = (const float*)d_in[18];
  const float* g_sc2  = (const float*)d_in[19];
  const float* be_sc2 = (const float*)d_in[20];

  float* out     = (float*)d_out;
  float* outXyz  = out;                               // [B,M,3]
  float* outY    = out + (size_t)BB * MM * 3;         // [B,D,M]
  int*   outSidx = (int*)(out + (size_t)BB * MM * 3 + (size_t)BB * DD * MM);

  char* ws = (char*)d_ws;
  size_t off = 0;
  auto take = [&](size_t bytes) -> char* {
    char* p = ws + off;
    off = (off + bytes + 255) & ~(size_t)255;
    return p;
  };
  int*      knnIdx  = (int*)take((size_t)BB * MM * KK * 4);
  float*    U       = (float*)take((size_t)BB * MM * KK * DD * 4);
  float*    meanF   = (float*)take((size_t)BB * MM * CC * 4);
  float*    centF   = (float*)take((size_t)BB * MM * CC * 4);
  float*    pooled  = (float*)take((size_t)BB * MM * DD * 4);
  float*    O       = (float*)take((size_t)BB * MM * DD * 4);
  float*    S1      = (float*)take((size_t)BB * MM * DD * 4);
  float*    S2      = (float*)take((size_t)BB * MM * DD * 4);
  float*    statsU  = (float*)take(128 * 4);
  float*    stats3  = (float*)take(384 * 4);
  float*    bnU     = (float*)take(128 * 4);
  float*    bn3     = (float*)take(384 * 4);
  _Float16* whScore = (_Float16*)take(4096 * 2);
  _Float16* whOut   = (_Float16*)take(4096 * 2);
  _Float16* whSc1   = (_Float16*)take(2048 * 2);
  _Float16* whSc2   = (_Float16*)take(2048 * 2);

  prep_kernel<<<16, 256, 0, stream>>>(W_score, W_out, W_sc1, W_sc2,
                                      whScore, whOut, whSc1, whSc2, statsU, stats3);

  knn_kernel<<<(BB * MM) / QPB, 32 * QPB, 0, stream>>>(xyz, sampleIdx, knnIdx,
                                                       outXyz, outSidx);

  stage1_kernel<<<(BB * MM * KK) / 256, 256, 0, stream>>>(
      xyz, feats, sampleIdx, knnIdx, W_pos, b_pos, W_feat, b_feat, U, statsU);

  scmean_kernel<<<(BB * MM * CC) / 256, 256, 0, stream>>>(feats, sampleIdx, knnIdx,
                                                          meanF, centF);

  finalizeU_kernel<<<1, 64, 0, stream>>>(statsU, g_feat, be_feat, g_pos, be_pos, bnU);

  attn_kernel<<<(BB * MM) / 4, 128, 0, stream>>>(U, bnU, whScore, pooled);

  gemm_stats_kernel<64><<<(BB * MM / 16) / 4, 128, 0, stream>>>(pooled, whOut, O, stats3);
  gemm_stats_kernel<32><<<(BB * MM / 16) / 4, 128, 0, stream>>>(meanF, whSc1, S1, stats3 + 128);
  gemm_stats_kernel<32><<<(BB * MM / 16) / 4, 128, 0, stream>>>(centF, whSc2, S2, stats3 + 256);

  finalize3_kernel<<<1, 192, 0, stream>>>(stats3, g_out, be_out, g_sc1, be_sc1,
                                          g_sc2, be_sc2, bn3);

  final_kernel<<<(BB * MM * DD) / 256, 256, 0, stream>>>(O, S1, S2, bn3, outY);
}
